// AttentionPooling_82480551953022
// MI455X (gfx1250) — matrix operand order repaired and checked by reading, not compiled, not run on hardware
//
#include <hip/hip_runtime.h>
#include <math.h>

typedef __attribute__((ext_vector_type(16))) _Float16 v16h;
typedef __attribute__((ext_vector_type(8)))  float    v8f;

#define D_IN  256
#define D_HID 128
#define KT 8                 // 256 / 32 K-tiles
#define NT 8                 // 128 / 16 N-tiles
#define SEG_ROWS 128

// ---------------------------------------------------------------------------
// Kernel 0: pack W1 (fp32 [256,128]) into f16 B-fragment layout.
// Linear slot t = kt*256 + nt*32 + lane holds the 16 halves lane `lane`
// feeds to v_wmma_f32_16x16x32_f16 for tile (kt,nt):
//   column n = nt*16 + (lane&15);  K = kt*32 + (lane<16 ? 0 : 16) + i
// ---------------------------------------------------------------------------
__global__ __launch_bounds__(256) void pack_w1_kernel(const float* __restrict__ W1,
                                                      _Float16* __restrict__ out) {
  int t = blockIdx.x * 256 + threadIdx.x;
  if (t >= KT * NT * 32) return;
  int lane = t & 31;
  int nt   = (t >> 5) & 7;
  int kt   = t >> 8;
  int col  = nt * 16 + (lane & 15);
  int kb   = kt * 32 + ((lane < 16) ? 0 : 16);
  _Float16* dst = out + (size_t)t * 16;
#pragma unroll
  for (int i = 0; i < 16; ++i)
    dst[i] = (_Float16)W1[(size_t)(kb + i) * D_HID + col];
}

// ---------------------------------------------------------------------------
// Kernel 1: fused Linear->ReLU->Linear producing scores[N].
// One wave per 16-row tile of x; 64 WMMAs per tile; W1 staged in 64KB LDS.
// ---------------------------------------------------------------------------
__global__ __launch_bounds__(256) void score_kernel(const float* __restrict__ x,
                                                    const _Float16* __restrict__ w1p,
                                                    const float* __restrict__ b1,
                                                    const float* __restrict__ W2,
                                                    const float* __restrict__ b2,
                                                    float* __restrict__ scores,
                                                    int n) {
  __shared__ _Float16 ldsW[KT * NT * 32 * 16];   // 32768 halves = 64 KB
  {
    const uint4* src = (const uint4*)w1p;
    uint4*       dst = (uint4*)ldsW;
    for (int i = threadIdx.x; i < (KT * NT * 32 * 16 * 2) / 16; i += 256)
      dst[i] = src[i];
  }
  __syncthreads();

  int wave = threadIdx.x >> 5;
  int lane = threadIdx.x & 31;
  int tile = blockIdx.x * 8 + wave;
  if (tile * 16 >= n) return;

  int row = tile * 16 + (lane & 15);
  if (row >= n) row = n - 1;                      // clamp (N is a multiple of 16 anyway)
  const float* xrow = x + (size_t)row * D_IN;
  int koffA = (lane < 16) ? 0 : 16;               // 16-bit A layout: half-wave K split (contiguous, matches B pack)

  v8f acc[NT];
  v8f zero = {0.f, 0.f, 0.f, 0.f, 0.f, 0.f, 0.f, 0.f};
#pragma unroll
  for (int ntc = 0; ntc < NT; ++ntc) acc[ntc] = zero;

  const v16h* bfrag = (const v16h*)ldsW;
#pragma unroll
  for (int kt = 0; kt < KT; ++kt) {
    int kb = kt * 32 + koffA;
    float4 f0 = *(const float4*)(xrow + kb);
    float4 f1 = *(const float4*)(xrow + kb + 4);
    float4 f2 = *(const float4*)(xrow + kb + 8);
    float4 f3 = *(const float4*)(xrow + kb + 12);
    v16h a;
    a[0]  = (_Float16)f0.x; a[1]  = (_Float16)f0.y; a[2]  = (_Float16)f0.z; a[3]  = (_Float16)f0.w;
    a[4]  = (_Float16)f1.x; a[5]  = (_Float16)f1.y; a[6]  = (_Float16)f1.z; a[7]  = (_Float16)f1.w;
    a[8]  = (_Float16)f2.x; a[9]  = (_Float16)f2.y; a[10] = (_Float16)f2.z; a[11] = (_Float16)f2.w;
    a[12] = (_Float16)f3.x; a[13] = (_Float16)f3.y; a[14] = (_Float16)f3.z; a[15] = (_Float16)f3.w;
#pragma unroll
    for (int ntc = 0; ntc < NT; ++ntc) {
      v16h b = bfrag[(kt * NT + ntc) * 32 + lane];
      acc[ntc] = __builtin_amdgcn_wmma_f32_16x16x32_f16(
          /*neg_a=*/false, a, /*neg_b=*/false, b,
          /*c_mod=*/(short)0, acc[ntc], /*reuse_a=*/false, /*reuse_b=*/false);
    }
  }

  // Epilogue: + b1, ReLU, dot with W2 (fused second linear).
  // C layout: VGPR r, lanes 0-15 -> row r, lanes 16-31 -> row r+8; column = nt*16 + (lane&15)
  int lc = lane & 15;
  float s[8];
#pragma unroll
  for (int r = 0; r < 8; ++r) s[r] = 0.f;
#pragma unroll
  for (int ntc = 0; ntc < NT; ++ntc) {
    int jj = ntc * 16 + lc;
    float bb = b1[jj];
    float ww = W2[jj];
#pragma unroll
    for (int r = 0; r < 8; ++r) {
      float h = acc[ntc][r] + bb;
      h = fmaxf(h, 0.f);
      s[r] += h * ww;
    }
  }
  // reduce across the 16 lanes sharing each row (stays within half-wave)
#pragma unroll
  for (int r = 0; r < 8; ++r) {
    s[r] += __shfl_xor(s[r], 1, 32);
    s[r] += __shfl_xor(s[r], 2, 32);
    s[r] += __shfl_xor(s[r], 4, 32);
    s[r] += __shfl_xor(s[r], 8, 32);
  }
  if (lc == 0) {                                   // lanes 0 (rows 0-7) and 16 (rows 8-15)
    float bias2 = b2[0];
    int base = tile * 16 + ((lane >> 4) << 3);
#pragma unroll
    for (int r = 0; r < 8; ++r)
      if (base + r < n) scores[base + r] = s[r] + bias2;
  }
}

// ---------------------------------------------------------------------------
// Kernel 2: global softmax statistics (max, sum of exp) — single 1024-thr block
// ---------------------------------------------------------------------------
__global__ __launch_bounds__(1024) void softmax_stats_kernel(const float* __restrict__ scores,
                                                             int n,
                                                             float* __restrict__ stats) {
  __shared__ float red[1024];
  int tid = threadIdx.x;
  float mx = -3.402823466e38f;
  for (int i = tid; i < n; i += 1024) mx = fmaxf(mx, scores[i]);
  red[tid] = mx;
  __syncthreads();
  for (int s = 512; s > 0; s >>= 1) {
    if (tid < s) red[tid] = fmaxf(red[tid], red[tid + s]);
    __syncthreads();
  }
  float gmax = red[0];
  __syncthreads();
  float sum = 0.f;
  for (int i = tid; i < n; i += 1024) sum += __expf(scores[i] - gmax);
  red[tid] = sum;
  __syncthreads();
  for (int s = 512; s > 0; s >>= 1) {
    if (tid < s) red[tid] += red[tid + s];
    __syncthreads();
  }
  if (tid == 0) { stats[0] = gmax; stats[1] = red[0]; }
}

// ---------------------------------------------------------------------------
// Kernel 3: zero pooled output (empty segments must be zero; d_out is poisoned)
// ---------------------------------------------------------------------------
__global__ void zero_kernel(float* __restrict__ p, int n) {
  int i = blockIdx.x * blockDim.x + threadIdx.x;
  if (i < n) p[i] = 0.f;
}

// ---------------------------------------------------------------------------
// Kernel 4: weighted segment-sum. One wave per 128 sorted rows; register
// accumulation over runs of equal batch id; atomic flush at boundaries only.
// ---------------------------------------------------------------------------
__global__ __launch_bounds__(256) void segsum_kernel(const float* __restrict__ x,
                                                     const long long* __restrict__ batch,
                                                     const float* __restrict__ scores,
                                                     const float* __restrict__ stats,
                                                     float* __restrict__ pooled,
                                                     int n) {
  long long wave = (long long)blockIdx.x * 8 + (threadIdx.x >> 5);
  int lane = threadIdx.x & 31;
  long long r0 = wave * SEG_ROWS;
  if (r0 >= n) return;
  long long r1 = r0 + SEG_ROWS;
  if (r1 > n) r1 = n;

  float gmax = stats[0];
  float inv  = 1.0f / stats[1];
  int col = lane * 8;                              // 32 lanes x 8 cols = 256

  float acc[8];
#pragma unroll
  for (int i = 0; i < 8; ++i) acc[i] = 0.f;
  long long cur = batch[r0];

  for (long long r = r0; r < r1; ++r) {
    long long b = batch[r];
    if (b != cur) {
      float* dst = pooled + (size_t)cur * D_IN + col;
#pragma unroll
      for (int i = 0; i < 8; ++i) { atomicAdd(dst + i, acc[i]); acc[i] = 0.f; }
      cur = b;
    }
    float w = __expf(scores[r] - gmax) * inv;
    const float4* xp = (const float4*)(x + (size_t)r * D_IN + col);
    float4 v0 = xp[0], v1 = xp[1];
    acc[0] += w * v0.x; acc[1] += w * v0.y; acc[2] += w * v0.z; acc[3] += w * v0.w;
    acc[4] += w * v1.x; acc[5] += w * v1.y; acc[6] += w * v1.z; acc[7] += w * v1.w;
  }
  float* dst = pooled + (size_t)cur * D_IN + col;
#pragma unroll
  for (int i = 0; i < 8; ++i) atomicAdd(dst + i, acc[i]);
}

// ---------------------------------------------------------------------------
extern "C" void kernel_launch(void* const* d_in, const int* in_sizes, int n_in,
                              void* d_out, int out_size, void* d_ws, size_t ws_size,
                              hipStream_t stream) {
  const float*     x     = (const float*)d_in[0];
  const long long* batch = (const long long*)d_in[1];   // jnp.int64, sorted
  const float*     W1    = (const float*)d_in[3];
  const float*     b1    = (const float*)d_in[4];
  const float*     W2    = (const float*)d_in[5];
  const float*     b2    = (const float*)d_in[6];
  float*           pooled = (float*)d_out;

  int n = in_sizes[0] / D_IN;                           // 300000

  // workspace layout
  char*     ws     = (char*)d_ws;
  _Float16* w1p    = (_Float16*)ws;                     // 64 KB packed W1
  float*    scores = (float*)(ws + 65536);              // n floats
  size_t    soff   = 65536 + (((size_t)n * 4 + 255) & ~(size_t)255);
  float*    stats  = (float*)(ws + soff);               // [max, sumexp]

  // 0: pack W1 -> f16 B-fragment layout
  pack_w1_kernel<<<(KT * NT * 32 + 255) / 256, 256, 0, stream>>>(W1, w1p);

  // 3 (independent of 0-2): zero output
  zero_kernel<<<(out_size + 255) / 256, 256, 0, stream>>>(pooled, out_size);

  // 1: WMMA MLP -> scores
  int tiles  = (n + 15) / 16;
  int blocks = (tiles + 7) / 8;
  score_kernel<<<blocks, 256, 0, stream>>>(x, w1p, b1, W2, b2, scores, n);

  // 2: softmax stats
  softmax_stats_kernel<<<1, 1024, 0, stream>>>(scores, n, stats);

  // 4: weighted segment sum
  long long waves = ((long long)n + SEG_ROWS - 1) / SEG_ROWS;
  int sblocks = (int)((waves + 7) / 8);
  segsum_kernel<<<sblocks, 256, 0, stream>>>(x, batch, scores, stats, pooled, n);
}
